// Model_13709535609460
// MI455X (gfx1250) — matrix operand order repaired and checked by reading, compiled, not run on hardware
//
#include <hip/hip_runtime.h>
#include <hip/hip_bf16.h>

typedef __attribute__((ext_vector_type(16))) _Float16 v16h;
typedef __attribute__((ext_vector_type(8)))  float    v8f;

#define N_NODES 50000
#define F_IN    64
#define P_IN    2
#define DIN     66      // F+P
#define H1      128
#define OUT     64
#define K1      160     // 2*DIN=132 padded up to multiple of 32
#define KT1     5       // K1/32
#define KT2     4       // H1/32
#define MTILES  3125    // 50000/16 (exact)
#define NT1     8       // H1/16
#define NT2     8       // 128/16 (troot|tnbr fused)

// ---------------- degree ----------------
__global__ void k_deg(const int* __restrict__ dst, float* __restrict__ deg, int E) {
    int e = blockIdx.x * blockDim.x + threadIdx.x;
    if (e < E) atomicAdd(&deg[dst[e]], 1.0f);
}
__global__ void k_deginv(float* __restrict__ deg) {
    int i = blockIdx.x * blockDim.x + threadIdx.x;
    if (i < N_NODES) deg[i] = 1.0f / fmaxf(deg[i], 1.0f);   // in place -> deg_inv
}

// ---------------- layer-1 neighbor scatter: agg0[dst] += h0[src], 66 dims ----------------
__global__ void k_scatter0(const int* __restrict__ src, const int* __restrict__ dst,
                           const float* __restrict__ x, const float* __restrict__ pos,
                           float* __restrict__ agg0, int E) {
    int lane = threadIdx.x & 31;
    int e = blockIdx.x * 8 + (threadIdx.x >> 5);
    if (e >= E) return;
    int s = src[e], d = dst[e];
    atomicAdd(&agg0[(size_t)d * DIN + lane],        x[(size_t)s * F_IN + lane]);
    atomicAdd(&agg0[(size_t)d * DIN + lane + 32],   x[(size_t)s * F_IN + lane + 32]);
    if (lane < P_IN)
        atomicAdd(&agg0[(size_t)d * DIN + F_IN + lane], pos[(size_t)s * P_IN + lane]);
}

// ---------------- build A1 (f16, WMMA-A swizzled): cols 0..65 h0, 66..131 deg_inv*agg0 ----------------
__global__ void k_build_A1(const float* __restrict__ x, const float* __restrict__ pos,
                           const float* __restrict__ agg0, const float* __restrict__ dinv,
                           _Float16* __restrict__ A1) {
    size_t idx = (size_t)blockIdx.x * blockDim.x + threadIdx.x;   // over N*K1 = 8,000,000
    int j  = idx & 15;
    int l  = (idx >> 4) & 31;
    int tk = (int)(idx >> 9);
    int kt = tk % KT1, mt = tk / KT1;
    int m = mt * 16 + (l & 15);
    int khalf = l >> 4;
    int k = kt * 32 + (j < 8 ? khalf * 8 + j : 16 + khalf * 8 + (j - 8));
    float v;
    if (k < F_IN)            v = x[(size_t)m * F_IN + k];
    else if (k < DIN)        v = pos[(size_t)m * P_IN + (k - F_IN)];
    else if (k < 2 * DIN)    v = dinv[m] * agg0[(size_t)m * DIN + (k - DIN)];
    else                     v = 0.0f;
    A1[idx] = (_Float16)v;
}

// ---------------- pack B1 (f16, WMMA-B swizzled): rows [W1_root; W1_nbr; 0] -> [K1, H1] ----------------
__global__ void k_pack_B1(const float* __restrict__ Wr, const float* __restrict__ Wn,
                          _Float16* __restrict__ B1) {
    int idx = blockIdx.x * blockDim.x + threadIdx.x;   // NT1*KT1*32*16 = 20480
    int j  = idx & 15;
    int l  = (idx >> 4) & 31;
    int tk = idx >> 9;
    int kt = tk % KT1, nt = tk / KT1;
    int n = nt * 16 + (l & 15);
    int k = kt * 32 + (j < 8 ? (l >> 4) * 8 + j : 16 + (l >> 4) * 8 + (j - 8));
    float v;
    if (k < DIN)           v = Wr[(size_t)k * H1 + n];
    else if (k < 2 * DIN)  v = Wn[(size_t)(k - DIN) * H1 + n];
    else                   v = 0.0f;
    B1[idx] = (_Float16)v;
}

// ---------------- pack B2 (f16): cols 0..63 W2_root, 64..127 W2_nbr -> [H1, 128] ----------------
__global__ void k_pack_B2(const float* __restrict__ Wr, const float* __restrict__ Wn,
                          _Float16* __restrict__ B2) {
    int idx = blockIdx.x * blockDim.x + threadIdx.x;   // NT2*KT2*32*16 = 16384
    int j  = idx & 15;
    int l  = (idx >> 4) & 31;
    int tk = idx >> 9;
    int kt = tk % KT2, nt = tk / KT2;
    int n = nt * 16 + (l & 15);
    int k = kt * 32 + (j < 8 ? (l >> 4) * 8 + j : 16 + (l >> 4) * 8 + (j - 8));
    float v = (n < OUT) ? Wr[(size_t)k * OUT + n] : Wn[(size_t)k * OUT + (n - OUT)];
    B2[idx] = (_Float16)v;
}

// ---------------- GEMM1: [N,160]x[160,128] + b1, ReLU -> h1 (f16, A-swizzled for GEMM2) ----------------
__global__ __launch_bounds__(256) void k_gemm1(const _Float16* __restrict__ A,
                                               const _Float16* __restrict__ B,
                                               const float* __restrict__ b1,
                                               _Float16* __restrict__ h1) {
    int lane = threadIdx.x & 31;
    int mt = blockIdx.x * 8 + (threadIdx.x >> 5);
    if (mt >= MTILES) return;
    v8f z = {};
    v8f acc[NT1];
#pragma unroll
    for (int nt = 0; nt < NT1; ++nt) acc[nt] = z;
#pragma unroll
    for (int kt = 0; kt < KT1; ++kt) {
        v16h a = *(const v16h*)(A + ((size_t)(mt * KT1 + kt) * 32 + lane) * 16);
        if (kt + 1 < KT1)
            __builtin_prefetch(A + ((size_t)(mt * KT1 + kt + 1) * 32 + lane) * 16, 0, 1);
#pragma unroll
        for (int nt = 0; nt < NT1; ++nt) {
            v16h b = *(const v16h*)(B + ((size_t)(nt * KT1 + kt) * 32 + lane) * 16);
            acc[nt] = __builtin_amdgcn_wmma_f32_16x16x32_f16(
                false, a, false, b, (short)0, acc[nt], false, false);
        }
    }
    int mlo = lane & 15, mhi = lane >> 4;
#pragma unroll
    for (int nt = 0; nt < NT1; ++nt) {
#pragma unroll
        for (int r = 0; r < 8; ++r) {
            int row = mt * 16 + r + 8 * mhi;
            int col = nt * 16 + mlo;            // k-index of next GEMM
            float v = acc[nt][r] + b1[col];
            v = v > 0.0f ? v : 0.0f;
            // store into A-swizzled layout for GEMM2 (K=128)
            int kt2 = col >> 5, kin = col & 31;
            int lhi, jj;
            if (kin < 16) { lhi = kin >> 3; jj = kin & 7; }
            else          { lhi = (kin - 16) >> 3; jj = 8 + ((kin - 16) & 7); }
            int l2 = (row & 15) + 16 * lhi;
            size_t o = ((size_t)((row >> 4) * KT2 + kt2) * 32 + l2) * 16 + jj;
            h1[o] = (_Float16)v;
        }
    }
}

// ---------------- GEMM2: [N,128]x[128,128] -> troot [N,64] | tnbr [N,64] (f32) ----------------
__global__ __launch_bounds__(256) void k_gemm2(const _Float16* __restrict__ A,
                                               const _Float16* __restrict__ B,
                                               float* __restrict__ troot,
                                               float* __restrict__ tnbr) {
    int lane = threadIdx.x & 31;
    int mt = blockIdx.x * 8 + (threadIdx.x >> 5);
    if (mt >= MTILES) return;
    v8f z = {};
    v8f acc[NT2];
#pragma unroll
    for (int nt = 0; nt < NT2; ++nt) acc[nt] = z;
#pragma unroll
    for (int kt = 0; kt < KT2; ++kt) {
        v16h a = *(const v16h*)(A + ((size_t)(mt * KT2 + kt) * 32 + lane) * 16);
#pragma unroll
        for (int nt = 0; nt < NT2; ++nt) {
            v16h b = *(const v16h*)(B + ((size_t)(nt * KT2 + kt) * 32 + lane) * 16);
            acc[nt] = __builtin_amdgcn_wmma_f32_16x16x32_f16(
                false, a, false, b, (short)0, acc[nt], false, false);
        }
    }
    int mlo = lane & 15, mhi = lane >> 4;
#pragma unroll
    for (int nt = 0; nt < NT2; ++nt) {
#pragma unroll
        for (int r = 0; r < 8; ++r) {
            int row = mt * 16 + r + 8 * mhi;
            int col = nt * 16 + mlo;
            float v = acc[nt][r];
            if (col < OUT) troot[(size_t)row * OUT + col] = v;
            else           tnbr[(size_t)row * OUT + (col - OUT)] = v;
        }
    }
}

// ---------------- layer-2 scatter on projected 64-dim features ----------------
__global__ void k_scatter2(const int* __restrict__ src, const int* __restrict__ dst,
                           const float* __restrict__ tnbr, float* __restrict__ agg2, int E) {
    int lane = threadIdx.x & 31;
    int e = blockIdx.x * 8 + (threadIdx.x >> 5);
    if (e >= E) return;
    int s = src[e], d = dst[e];
    atomicAdd(&agg2[(size_t)d * OUT + lane],      tnbr[(size_t)s * OUT + lane]);
    atomicAdd(&agg2[(size_t)d * OUT + lane + 32], tnbr[(size_t)s * OUT + lane + 32]);
}

// ---------------- combine: h2 = troot + deg_inv*agg2 + b2 ----------------
__global__ void k_combine(const float* __restrict__ troot, const float* __restrict__ agg2,
                          const float* __restrict__ dinv, const float* __restrict__ b2,
                          float* __restrict__ h2) {
    size_t idx = (size_t)blockIdx.x * blockDim.x + threadIdx.x;   // N*64
    int i = (int)(idx >> 6), f = (int)(idx & 63);
    h2[idx] = troot[idx] + dinv[i] * agg2[idx] + b2[f];
}

// ---------------- per-edge dot product ----------------
__global__ void k_edge_dot(const int* __restrict__ src, const int* __restrict__ dst,
                           const float* __restrict__ h2, float* __restrict__ out, int E) {
    int lane = threadIdx.x & 31;
    int e = blockIdx.x * 8 + (threadIdx.x >> 5);
    if (e >= E) return;
    int s = src[e], d = dst[e];
    float p = h2[(size_t)s * OUT + lane]      * h2[(size_t)d * OUT + lane]
            + h2[(size_t)s * OUT + lane + 32] * h2[(size_t)d * OUT + lane + 32];
#pragma unroll
    for (int off = 16; off > 0; off >>= 1) p += __shfl_xor(p, off, 32);
    if (lane == 0) out[e] = p;
}

// ---------------- workspace layout (bytes, 256-aligned; aliased regions) ----------------
#define OFF_DEG   ((size_t)0)            // N f32 deg -> deg_inv (in place)      200,192
#define OFF_AGG   ((size_t)200192)       // agg0 [N,66] f32 ; later agg2 [N,64]  13,200,128
#define OFF_A1    ((size_t)13400320)     // A1 swizzled f16 [N,160]; later h2    16,000,000
#define OFF_H1    ((size_t)29400320)     // h1 swizzled f16 [N,128]              12,800,000
#define OFF_T     ((size_t)42200320)     // troot|tnbr f32 [N,128]               25,600,000
#define OFF_B1    ((size_t)67800320)     // B1 swizzled f16                      40,960
#define OFF_B2    ((size_t)67841280)     // B2 swizzled f16                      32,768

extern "C" void kernel_launch(void* const* d_in, const int* in_sizes, int n_in,
                              void* d_out, int out_size, void* d_ws, size_t ws_size,
                              hipStream_t stream) {
    const float* x       = (const float*)d_in[0];
    const float* pos     = (const float*)d_in[1];
    const int*   ei      = (const int*)d_in[2];
    const float* W1_root = (const float*)d_in[3];
    const float* W1_nbr  = (const float*)d_in[4];
    const float* b1      = (const float*)d_in[5];
    const float* W2_root = (const float*)d_in[6];
    const float* W2_nbr  = (const float*)d_in[7];
    const float* b2      = (const float*)d_in[8];

    const int E = in_sizes[2] / 2;
    const int* src = ei;
    const int* dst = ei + E;

    char* ws = (char*)d_ws;
    float*     deg   = (float*)(ws + OFF_DEG);
    float*     agg0  = (float*)(ws + OFF_AGG);
    float*     agg2  = (float*)(ws + OFF_AGG);      // alias: agg0 dead after build_A1
    _Float16*  A1    = (_Float16*)(ws + OFF_A1);
    float*     h2    = (float*)(ws + OFF_A1);       // alias: A1 dead after gemm1
    _Float16*  h1    = (_Float16*)(ws + OFF_H1);
    float*     troot = (float*)(ws + OFF_T);
    float*     tnbr  = (float*)(ws + OFF_T) + (size_t)N_NODES * OUT;
    _Float16*  B1    = (_Float16*)(ws + OFF_B1);
    _Float16*  B2    = (_Float16*)(ws + OFF_B2);

    const int eb  = (E + 255) / 256;          // thread-per-edge blocks
    const int ewb = (E + 7) / 8;              // wave-per-edge blocks (256 thr = 8 waves)
    const int gmm = (MTILES + 7) / 8;         // gemm blocks

    // degree / deg_inv
    hipMemsetAsync(deg, 0, (size_t)N_NODES * 4, stream);
    k_deg<<<eb, 256, 0, stream>>>(dst, deg, E);
    k_deginv<<<(N_NODES + 255) / 256, 256, 0, stream>>>(deg);

    // layer 1 aggregation + operand packing
    hipMemsetAsync(agg0, 0, (size_t)N_NODES * DIN * 4, stream);
    k_scatter0<<<ewb, 256, 0, stream>>>(src, dst, x, pos, agg0, E);
    k_build_A1<<<(N_NODES * K1) / 256, 256, 0, stream>>>(x, pos, agg0, deg, A1);
    k_pack_B1<<<(NT1 * KT1 * 32 * 16) / 256, 256, 0, stream>>>(W1_root, W1_nbr, B1);

    // layer 1 GEMM (WMMA) with bias + ReLU, writes h1 pre-swizzled
    k_gemm1<<<gmm, 256, 0, stream>>>(A1, B1, b1, h1);

    // layer 2 GEMM (WMMA): both projections at once
    k_pack_B2<<<(NT2 * KT2 * 32 * 16) / 256, 256, 0, stream>>>(W2_root, W2_nbr, B2);
    k_gemm2<<<gmm, 256, 0, stream>>>(h1, B2, troot, tnbr);

    // layer 2 aggregation on the 64-dim projected features (transform-then-aggregate)
    hipMemsetAsync(agg2, 0, (size_t)N_NODES * OUT * 4, stream);
    k_scatter2<<<ewb, 256, 0, stream>>>(src, dst, tnbr, agg2, E);
    k_combine<<<(N_NODES * OUT) / 256, 256, 0, stream>>>(troot, agg2, deg, b2, h2);

    // edge-wise dot product
    k_edge_dot<<<ewb, 256, 0, stream>>>(src, dst, h2, (float*)d_out, E);
}
